// FeatureExtractionBlock_PNA_23776938951298
// MI455X (gfx1250) — compile-verified, matrix operand order checked by reading, not run on hardware
//
#include <hip/hip_runtime.h>
#include <hip/hip_bf16.h>

typedef _Float16 half_t;
typedef __attribute__((ext_vector_type(8)))  _Float16 v8h;
typedef __attribute__((ext_vector_type(16))) _Float16 v16h;
typedef __attribute__((ext_vector_type(8)))  float    v8f;

#define NTOK 32768   // H*W*Z = 64*64*8
#define CCH  128
#define HH   64
#define WW   64
#define ZZ   8
#define HEADS 8
#define HD    16

// ---------------------------------------------------------------------------
// Weight prep: fp32 [K,N] -> f16 transposed [N,K] so B-fragments load exactly
// like A-fragments (per-lane contiguous 8-half runs).
// ---------------------------------------------------------------------------
__global__ __launch_bounds__(256) void prep_weights(
    const float* __restrict__ qkv_w, const float* __restrict__ proj_w,
    const float* __restrict__ w1,    const float* __restrict__ w2,
    _Float16* __restrict__ wq, _Float16* __restrict__ wp,
    _Float16* __restrict__ w1t, _Float16* __restrict__ w2t)
{
    int e = blockIdx.x * 256 + threadIdx.x;            // 0 .. 196607
    if (e < 49152)       { int n = e >> 7, k = e & 127;                 wq[e]  = (_Float16)qkv_w[k * 384 + n]; }
    else if (e < 65536)  { int i = e - 49152;  int n = i >> 7, k = i & 127; wp[i]  = (_Float16)proj_w[k * 128 + n]; }
    else if (e < 131072) { int i = e - 65536;  int n = i >> 7, k = i & 127; w1t[i] = (_Float16)w1[k * 512 + n]; }
    else                 { int i = e - 131072; int n = i >> 9, k = i & 511; w2t[i] = (_Float16)w2[k * 128 + n]; }
}

// ---------------------------------------------------------------------------
// LN1: read x in (C,H,W,Z) layout (coalesced over tokens), LayerNorm over C,
// emit token-major [N,128] f16 for the WMMA A-operand.
// ---------------------------------------------------------------------------
__global__ __launch_bounds__(128) void ln1_kernel(
    const float* __restrict__ x, const float* __restrict__ g,
    const float* __restrict__ b, _Float16* __restrict__ xn)
{
    __shared__ float tile[128 * 32];
    __shared__ float smu[32], srs[32];
    int tid = threadIdx.x;
    int token0 = blockIdx.x * 32;

    for (int idx = tid; idx < 4096; idx += 128) {
        int c = idx >> 5, t = idx & 31;
        tile[idx] = x[(size_t)c * NTOK + token0 + t];
    }
    __syncthreads();
    if (tid < 32) {
        float s = 0.f, s2 = 0.f;
        #pragma unroll 8
        for (int c = 0; c < 128; ++c) { float v = tile[c * 32 + tid]; s += v; s2 += v * v; }
        float mu  = s * (1.0f / 128.0f);
        float var = s2 * (1.0f / 128.0f) - mu * mu;
        smu[tid] = mu; srs[tid] = rsqrtf(var + 1e-5f);
    }
    __syncthreads();
    for (int idx = tid; idx < 4096; idx += 128) {
        int t = idx >> 7, c = idx & 127;
        float v = (tile[c * 32 + t] - smu[t]) * srs[t] * g[c] + b[c];
        xn[(size_t)(token0 + t) * 128 + c] = (_Float16)v;
    }
}

// ---------------------------------------------------------------------------
// LN2: x1 is token-major fp32; one 128-thread block per token.
// ---------------------------------------------------------------------------
__global__ __launch_bounds__(128) void ln2_kernel(
    const float* __restrict__ x1, const float* __restrict__ g,
    const float* __restrict__ b, _Float16* __restrict__ out)
{
    __shared__ float s1[128], s2[128];
    int token = blockIdx.x, c = threadIdx.x;
    float v = x1[(size_t)token * 128 + c];
    s1[c] = v; s2[c] = v * v;
    __syncthreads();
    for (int off = 64; off > 0; off >>= 1) {
        if (c < off) { s1[c] += s1[c + off]; s2[c] += s2[c + off]; }
        __syncthreads();
    }
    float mu  = s1[0] * (1.0f / 128.0f);
    float var = s2[0] * (1.0f / 128.0f) - mu * mu;
    float r = rsqrtf(var + 1e-5f);
    out[(size_t)token * 128 + c] = (_Float16)((v - mu) * r * g[c] + b[c]);
}

// ---------------------------------------------------------------------------
// Register-blocked WMMA GEMM: each wave computes a 32x64 output tile with
// 8 f32 accumulators; per K=32 step it loads 2 A-frags + 4 B-frags (12 b128)
// and issues 8 V_WMMA_F32_16X16X32_F16 -> 1.5 loads per WMMA.
// Fragment layout per ISA: lane l<16 holds row l, K = {0..7,16..23};
// lane l+16 holds row l, K = {8..15,24..31}. BT is [Nc,K] so B loads mirror A.
// ---------------------------------------------------------------------------
__device__ inline v16h load_frag16(const _Float16* p)
{
    v8h lo = *(const v8h*)(p);
    v8h hi = *(const v8h*)(p + 16);
    return __builtin_shufflevector(lo, hi, 0,1,2,3,4,5,6,7,8,9,10,11,12,13,14,15);
}

__device__ inline float gelu_tanh(float v)
{
    float t = tanhf(0.7978845608028654f * (v + 0.044715f * v * v * v));
    return 0.5f * v * (1.0f + t);
}

#define WMMA_F16(A_, B_, C_) __builtin_amdgcn_wmma_f32_16x16x32_f16( \
    false, (A_), false, (B_), (short)0, (C_), false, false)

__global__ __launch_bounds__(256) void gemm_wmma(
    const _Float16* __restrict__ A, const _Float16* __restrict__ BT,
    const float* __restrict__ bias, void* __restrict__ outp,
    const float* __restrict__ resid,
    int M, int Nc, int K, int residColMajor, int actGelu, int outF16,
    int outColMajor)
{
    int lane = threadIdx.x & 31;
    int wave = threadIdx.x >> 5;
    int tilesN = Nc >> 6;                                // 64-wide N tiles
    int tile = blockIdx.x * (blockDim.x >> 5) + wave;
    if (tile >= (M >> 5) * tilesN) return;               // wave-uniform exit
    int tm = tile / tilesN, tn = tile - tm * tilesN;
    int l = lane & 15, hf = lane >> 4;

    const _Float16* aBase = A  + (size_t)(tm * 32 + l) * K + hf * 8;
    const _Float16* bBase = BT + (size_t)(tn * 64 + l) * K + hf * 8;
    const size_t aStep = (size_t)16 * K;                 // 16 rows
    const size_t bStep = (size_t)16 * K;                 // 16 cols

    v8f acc0 = {}, acc1 = {}, acc2 = {}, acc3 = {};
    v8f acc4 = {}, acc5 = {}, acc6 = {}, acc7 = {};

    for (int kk = 0; kk < K; kk += 32) {
        __builtin_prefetch(aBase + kk + 32, 0, 1);       // global_prefetch_b8
        __builtin_prefetch(bBase + kk + 32, 0, 1);
        v16h a0 = load_frag16(aBase + kk);
        v16h a1 = load_frag16(aBase + aStep + kk);
        v16h b0 = load_frag16(bBase + kk);
        v16h b1 = load_frag16(bBase + bStep     + kk);
        v16h b2 = load_frag16(bBase + bStep * 2 + kk);
        v16h b3 = load_frag16(bBase + bStep * 3 + kk);
        acc0 = WMMA_F16(a0, b0, acc0);
        acc1 = WMMA_F16(a0, b1, acc1);
        acc2 = WMMA_F16(a0, b2, acc2);
        acc3 = WMMA_F16(a0, b3, acc3);
        acc4 = WMMA_F16(a1, b0, acc4);
        acc5 = WMMA_F16(a1, b1, acc5);
        acc6 = WMMA_F16(a1, b2, acc6);
        acc7 = WMMA_F16(a1, b3, acc7);
    }

    v8f accs[8] = {acc0, acc1, acc2, acc3, acc4, acc5, acc6, acc7};
    #pragma unroll
    for (int i = 0; i < 2; ++i) {
        #pragma unroll
        for (int j = 0; j < 4; ++j) {
            int col = tn * 64 + j * 16 + l;
            float bv = bias[col];
            #pragma unroll
            for (int r = 0; r < 8; ++r) {
                int row = tm * 32 + i * 16 + hf * 8 + r;
                float v = accs[i * 4 + j][r] + bv;
                if (actGelu) v = gelu_tanh(v);
                if (resid) v += residColMajor ? resid[(size_t)col * M + row]
                                              : resid[(size_t)row * Nc + col];
                size_t oidx = outColMajor ? (size_t)col * M + row
                                          : (size_t)row * Nc + col;
                if (outF16) ((_Float16*)outp)[oidx] = (_Float16)v;
                else        ((float*)outp)   [oidx] = v;
            }
        }
    }
}

// ---------------------------------------------------------------------------
// 3D neighborhood attention (k=3, K3=27, hd=16): one thread per (token, head).
// ---------------------------------------------------------------------------
__global__ __launch_bounds__(256) void natten3d(
    const float* __restrict__ qkv, const float* __restrict__ rpb,
    _Float16* __restrict__ attn_out)
{
    int gid = blockIdx.x * 256 + threadIdx.x;
    int head = gid & 7;
    int token = gid >> 3;
    if (token >= NTOK) return;
    int z = token & 7, w = (token >> 3) & 63, h = token >> 9;
    int hs = h - 1; hs = hs < 0 ? 0 : (hs > HH - 3 ? HH - 3 : hs);
    int ws = w - 1; ws = ws < 0 ? 0 : (ws > WW - 3 ? WW - 3 : ws);
    int zs = z - 1; zs = zs < 0 ? 0 : (zs > ZZ - 3 ? ZZ - 3 : zs);

    const float* qp = qkv + (size_t)token * 384 + head * HD;
    float q[HD];
    #pragma unroll
    for (int e = 0; e < HD; ++e) q[e] = qp[e] * 0.25f;   // hd^-0.5

    float sc[27];
    float mx = -3.0e38f;
    #pragma unroll
    for (int m = 0; m < 27; ++m) {
        int dh = m / 9, dw = (m / 3) % 3, dz = m % 3;
        int nh = hs + dh, nw = ws + dw, nz = zs + dz;
        int ntok = ((nh << 6) + nw) * 8 + nz;
        const float* kp = qkv + (size_t)ntok * 384 + 128 + head * HD;
        float d = 0.f;
        #pragma unroll
        for (int e = 0; e < HD; ++e) d += q[e] * kp[e];
        int rh = nh - h + 2, rw = nw - w + 2, rz = nz - z + 2;
        d += rpb[(((head * 5) + rh) * 5 + rw) * 5 + rz];
        sc[m] = d;
        mx = fmaxf(mx, d);
    }
    float ssum = 0.f;
    #pragma unroll
    for (int m = 0; m < 27; ++m) { sc[m] = __expf(sc[m] - mx); ssum += sc[m]; }
    float inv = 1.0f / ssum;

    float o[HD];
    #pragma unroll
    for (int e = 0; e < HD; ++e) o[e] = 0.f;
    #pragma unroll
    for (int m = 0; m < 27; ++m) {
        int dh = m / 9, dw = (m / 3) % 3, dz = m % 3;
        int ntok = (((hs + dh) << 6) + (ws + dw)) * 8 + (zs + dz);
        const float* vp = qkv + (size_t)ntok * 384 + 256 + head * HD;
        float p = sc[m] * inv;
        #pragma unroll
        for (int e = 0; e < HD; ++e) o[e] += p * vp[e];
    }
    _Float16* op = attn_out + (size_t)token * 128 + head * HD;
    #pragma unroll
    for (int e = 0; e < HD; ++e) op[e] = (_Float16)o[e];
}

// ---------------------------------------------------------------------------
// Final fusion: skip = x2 + x (both (C,H,W,Z) layout, fully coalesced) and
// pooled = maxpool 2x2x2 over skip. One thread per (c, ph, pw) covers 2x2x8.
// x2 was written column-major by the MLP2 GEMM epilogue.
// ---------------------------------------------------------------------------
__global__ __launch_bounds__(256) void skip_pool(
    const float* __restrict__ x2cm, const float* __restrict__ x,
    float* __restrict__ outp)
{
    int id = blockIdx.x * 256 + threadIdx.x;    // < 128*32*32
    int c = id >> 10;
    int rem = id & 1023;
    int ph = rem >> 5, pw = rem & 31;
    float* pooled = outp;                       // (128,32,32,4)
    float* skip   = outp + 524288;              // (128,64,64,8)

    float pv[4] = {-3.0e38f, -3.0e38f, -3.0e38f, -3.0e38f};
    #pragma unroll
    for (int dh = 0; dh < 2; ++dh)
    #pragma unroll
    for (int dw = 0; dw < 2; ++dw) {
        int hh = 2 * ph + dh, ww2 = 2 * pw + dw;
        int tb = (hh * 64 + ww2) * 8;
        const float* xin = x    + (size_t)c * NTOK + tb;
        const float* x2p = x2cm + (size_t)c * NTOK + tb;
        #pragma unroll
        for (int zc = 0; zc < 8; ++zc) {
            float s = x2p[zc] + xin[zc];
            skip[(size_t)c * NTOK + tb + zc] = s;
            pv[zc >> 1] = fmaxf(pv[zc >> 1], s);
        }
    }
    float* pp = pooled + (size_t)c * 4096 + ph * 128 + pw * 4;
    #pragma unroll
    for (int pz = 0; pz < 4; ++pz) pp[pz] = pv[pz];
}

// ---------------------------------------------------------------------------
extern "C" void kernel_launch(void* const* d_in, const int* in_sizes, int n_in,
                              void* d_out, int out_size, void* d_ws, size_t ws_size,
                              hipStream_t stream)
{
    const float* x      = (const float*)d_in[0];
    const float* ln1_g  = (const float*)d_in[1];
    const float* ln1_b  = (const float*)d_in[2];
    const float* qkv_w  = (const float*)d_in[3];
    const float* qkv_b  = (const float*)d_in[4];
    const float* proj_w = (const float*)d_in[5];
    const float* proj_b = (const float*)d_in[6];
    const float* rpb    = (const float*)d_in[7];
    const float* ln2_g  = (const float*)d_in[8];
    const float* ln2_b  = (const float*)d_in[9];
    const float* w1     = (const float*)d_in[10];
    const float* b1     = (const float*)d_in[11];
    const float* w2     = (const float*)d_in[12];
    const float* b2     = (const float*)d_in[13];

    char* ws = (char*)d_ws;
    _Float16* wqkvT  = (_Float16*)ws;                    // 49152 halfs
    _Float16* wprojT = wqkvT + 49152;                    // 16384
    _Float16* w1T    = wqkvT + 65536;                    // 65536
    _Float16* w2T    = wqkvT + 131072;                   // 65536  (ends @393216 B)
    _Float16* xn   = (_Float16*)(ws + 393216);           // [N,128] f16, 8 MB
    float*    qkv  = (float*)   (ws + 8781824);          // [N,384] f32, 48 MB
    _Float16* attn = (_Float16*)(ws + 59113472);         // [N,128] f16, 8 MB
    float*    x1   = (float*)   (ws + 67502080);         // [N,128] f32, 16 MB
    _Float16* x1n  = (_Float16*)(ws + 84279296);         // [N,128] f16, 8 MB
    _Float16* hbuf = (_Float16*)(ws + 92667904);         // [N,512] f16, 32 MB
    float*    x2   = (float*)   (ws + 126222336);        // [C,N] f32 (col-major), 16 MB

    prep_weights<<<768, 256, 0, stream>>>(qkv_w, proj_w, w1, w2, wqkvT, wprojT, w1T, w2T);
    ln1_kernel<<<NTOK / 32, 128, 0, stream>>>(x, ln1_g, ln1_b, xn);

    // qkv = LN(x) @ Wqkv + b : 1024 x 6 32x64-tiles = 6144 waves / 8 per block
    gemm_wmma<<<768, 256, 0, stream>>>(xn, wqkvT, qkv_b, qkv, nullptr,
                                       NTOK, 384, 128, 0, 0, 0, 0);
    natten3d<<<NTOK * HEADS / 256, 256, 0, stream>>>(qkv, rpb, attn);

    // x1 = x^T + attn @ Wp + bp : residual is column-major (C,HWZ) view of x
    gemm_wmma<<<256, 256, 0, stream>>>(attn, wprojT, proj_b, x1, x,
                                       NTOK, 128, 128, 1, 0, 0, 0);
    ln2_kernel<<<NTOK, 128, 0, stream>>>(x1, ln2_g, ln2_b, x1n);

    // h = gelu(LN(x1) @ W1 + b1) -> f16
    gemm_wmma<<<1024, 256, 0, stream>>>(x1n, w1T, b1, hbuf, nullptr,
                                        NTOK, 512, 128, 0, 1, 1, 0);
    // x2 = x1 + h @ W2 + b2, stored column-major (C,N) for the fused tail
    gemm_wmma<<<256, 256, 0, stream>>>(hbuf, w2T, b2, x2, x1,
                                       NTOK, 128, 512, 0, 0, 0, 1);

    skip_pool<<<512, 256, 0, stream>>>(x2, x, (float*)d_out);
}
